// attention_Block_8813272891665
// MI455X (gfx1250) — matrix operand order repaired and checked by reading, not compiled, not run on hardware
//
#include <hip/hip_runtime.h>
#include <math.h>

#define B_  4
#define N_  1024
#define D_  512
#define H_  8
#define DH_ 64
#define FF_ 2048
#define M_  (B_*N_)

// GEMM macro-tile
#define BM_ 128
#define BN_ 64
#define BK_ 32
#define LDA_ 40   // padded LDS stride (bf16 elems): 80B, 16B-aligned, conflict-free
#define LDB_ 40

typedef __attribute__((ext_vector_type(16))) __bf16 v16bf;
typedef __attribute__((ext_vector_type(8)))  float  v8f;

union BF16x16 {
  v16bf v;
  unsigned short s[16];
  unsigned int   u[8];
};

__device__ __forceinline__ unsigned short bfbits(float f) {
  return __builtin_bit_cast(unsigned short, (__bf16)f);   // hw v_cvt + bitcast
}

// ---------------------------------------------------------------------------
// Kernel 1: pe = coords@W_pe + b_pe ; t = x + pe ; q_in = LN(t, g_peq) ;
//           k_in = LN(t, g_pek)  (shared mean/var)
// ---------------------------------------------------------------------------
__global__ void pe_ln2_kernel(const float* __restrict__ x, const float* __restrict__ coords,
                              const float* __restrict__ Wpe, const float* __restrict__ bpe,
                              const float* __restrict__ gq, const float* __restrict__ bq,
                              const float* __restrict__ gk, const float* __restrict__ bk,
                              float* __restrict__ qout, float* __restrict__ kout) {
  int row = blockIdx.x;
  int tid = threadIdx.x;          // 256 threads, 2 elems each
  float c0 = coords[row * 2 + 0];
  float c1 = coords[row * 2 + 1];
  const float* xr = x + (size_t)row * D_;
  float t[2];
  float s = 0.f, s2 = 0.f;
#pragma unroll
  for (int e = 0; e < 2; ++e) {
    int d = tid + e * 256;
    float pe = c0 * Wpe[d] + c1 * Wpe[D_ + d] + bpe[d];
    float v = xr[d] + pe;
    t[e] = v; s += v; s2 += v * v;
  }
  __shared__ float rs[256], rs2[256];
  rs[tid] = s; rs2[tid] = s2; __syncthreads();
  for (int o = 128; o > 0; o >>= 1) {
    if (tid < o) { rs[tid] += rs[tid + o]; rs2[tid] += rs2[tid + o]; }
    __syncthreads();
  }
  float mean = rs[0] * (1.0f / D_);
  float var  = rs2[0] * (1.0f / D_) - mean * mean;
  float inv  = rsqrtf(var + 1e-5f);
#pragma unroll
  for (int e = 0; e < 2; ++e) {
    int d = tid + e * 256;
    float nv = (t[e] - mean) * inv;
    qout[(size_t)row * D_ + d] = nv * gq[d] + bq[d];
    kout[(size_t)row * D_ + d] = nv * gk[d] + bk[d];
  }
}

// ---------------------------------------------------------------------------
// Kernel 2: LDS-staged WMMA bf16 GEMM  Y = act(X[M,K] @ W[K,Nc] + bias)
// Block = 8 waves = 128x64 macro-tile; wave w owns rows w*16..w*16+15.
// Per K-step (32): coalesced global->bf16->LDS staging (B transposed),
// then fragments via ds_load_b128.  act: 0 = none, 1 = leaky_relu(0.2)
// ---------------------------------------------------------------------------
__global__ void gemm_wmma_kernel(const float* __restrict__ X, const float* __restrict__ W,
                                 const float* __restrict__ bias, float* __restrict__ Y,
                                 int M, int K, int Nc, int act) {
  __shared__ unsigned short sA[BM_ * LDA_];   // [row][k]   10240 B
  __shared__ unsigned short sB[BN_ * LDB_];   // [n][k] (transposed)  5120 B

  int tid  = threadIdx.x;
  int lane = tid & 31;
  int wid  = tid >> 5;
  int tiles_n = Nc >> 6;
  int bm = blockIdx.x / tiles_n;
  int bn = blockIdx.x % tiles_n;
  int row0 = bm * BM_, col0 = bn * BN_;
  int half = lane >> 4, l15 = lane & 15;

  // cooperative-load mapping
  int arow = tid >> 1;                 // 0..127
  int akc  = (tid & 1) << 4;           // 0 or 16
  int bkr  = tid >> 3;                 // 0..31
  int bnc  = (tid & 7) << 3;           // 0,8,..,56

  const float* ag = X + (size_t)(row0 + arow) * K + akc;
  const float* bg = W + (size_t)bkr * Nc + col0 + bnc;

  v8f c[4] = {};
  for (int k0 = 0; k0 < K; k0 += BK_) {
    // ---- stage A: 128x32 fp32 -> bf16 LDS ----
    {
      const float4* p = (const float4*)(ag + k0);
      float4 f0 = p[0], f1 = p[1], f2 = p[2], f3 = p[3];
      unsigned u0 = ((unsigned)bfbits(f0.y) << 16) | bfbits(f0.x);
      unsigned u1 = ((unsigned)bfbits(f0.w) << 16) | bfbits(f0.z);
      unsigned u2 = ((unsigned)bfbits(f1.y) << 16) | bfbits(f1.x);
      unsigned u3 = ((unsigned)bfbits(f1.w) << 16) | bfbits(f1.z);
      unsigned u4 = ((unsigned)bfbits(f2.y) << 16) | bfbits(f2.x);
      unsigned u5 = ((unsigned)bfbits(f2.w) << 16) | bfbits(f2.z);
      unsigned u6 = ((unsigned)bfbits(f3.y) << 16) | bfbits(f3.x);
      unsigned u7 = ((unsigned)bfbits(f3.w) << 16) | bfbits(f3.z);
      unsigned short* dst = sA + arow * LDA_ + akc;   // 16B-aligned
      *(uint4*)(dst)     = make_uint4(u0, u1, u2, u3);
      *(uint4*)(dst + 8) = make_uint4(u4, u5, u6, u7);
      __builtin_prefetch(ag + k0 + BK_, 0, 1);
    }
    // ---- stage B: 32x64 fp32 -> bf16 LDS, transposed [n][k] ----
    {
      const float4* p = (const float4*)(bg + (size_t)k0 * Nc);
      float4 f0 = p[0], f1 = p[1];
      sB[(bnc + 0) * LDB_ + bkr] = bfbits(f0.x);
      sB[(bnc + 1) * LDB_ + bkr] = bfbits(f0.y);
      sB[(bnc + 2) * LDB_ + bkr] = bfbits(f0.z);
      sB[(bnc + 3) * LDB_ + bkr] = bfbits(f0.w);
      sB[(bnc + 4) * LDB_ + bkr] = bfbits(f1.x);
      sB[(bnc + 5) * LDB_ + bkr] = bfbits(f1.y);
      sB[(bnc + 6) * LDB_ + bkr] = bfbits(f1.z);
      sB[(bnc + 7) * LDB_ + bkr] = bfbits(f1.w);
      __builtin_prefetch(bg + (size_t)(k0 + BK_) * Nc, 0, 1);
    }
    __syncthreads();
    // ---- compute: A frag (2x b128) + 4x { B frag (2x b128) ; WMMA } ----
    BF16x16 a;
    {
      const unsigned short* ap = sA + (wid * 16 + l15) * LDA_ + (half << 3);
      uint4 lo = *(const uint4*)(ap);        // K = kb..kb+7
      uint4 hi = *(const uint4*)(ap + 16);   // K = kb+16..kb+23
      a.u[0] = lo.x; a.u[1] = lo.y; a.u[2] = lo.z; a.u[3] = lo.w;
      a.u[4] = hi.x; a.u[5] = hi.y; a.u[6] = hi.z; a.u[7] = hi.w;
    }
#pragma unroll
    for (int n4 = 0; n4 < 4; ++n4) {
      const unsigned short* bp = sB + (n4 * 16 + l15) * LDB_ + (half << 3);
      uint4 lo = *(const uint4*)(bp);        // K = kb..kb+7
      uint4 hi = *(const uint4*)(bp + 16);   // K = kb+16..kb+23
      BF16x16 bb;
      bb.u[0] = lo.x; bb.u[1] = lo.y; bb.u[2] = lo.z; bb.u[3] = lo.w;
      bb.u[4] = hi.x; bb.u[5] = hi.y; bb.u[6] = hi.z; bb.u[7] = hi.w;
      c[n4] = __builtin_amdgcn_wmma_f32_16x16x32_bf16(false, a.v, false, bb.v,
                                                      (short)0, c[n4], false, false);
    }
    __syncthreads();
  }
  // ---- epilogue ----
  int rbase = (half << 3);
#pragma unroll
  for (int n4 = 0; n4 < 4; ++n4) {
    int col = col0 + n4 * 16 + l15;
    float bv = bias ? bias[col] : 0.f;
#pragma unroll
    for (int r = 0; r < 8; ++r) {
      float y = c[n4][r] + bv;
      if (act == 1) y = (y > 0.f) ? y : 0.2f * y;
      Y[(size_t)(row0 + wid * 16 + rbase + r) * Nc + col] = y;
    }
  }
}

// ---------------------------------------------------------------------------
// Kernel 3: per-(row,head) L2 normalize; optionally fold logit scale into q.
// ---------------------------------------------------------------------------
__global__ void headnorm_kernel(float* __restrict__ q, const float* __restrict__ ls,
                                int applyScale) {
  int tid = threadIdx.x;                 // 64
  int h   = blockIdx.x % H_;
  int row = blockIdx.x / H_;
  size_t idx = (size_t)row * D_ + h * DH_ + tid;
  float v = q[idx];
  __shared__ float red[64];
  red[tid] = v * v; __syncthreads();
  for (int o = 32; o > 0; o >>= 1) {
    if (tid < o) red[tid] += red[tid + o];
    __syncthreads();
  }
  float nrm = fmaxf(sqrtf(red[0]), 1e-12f);
  float sc = 1.f;
  if (applyScale) sc = __expf(fminf(ls[h], 4.60517019f));  // ln(100)
  q[idx] = v / nrm * sc;
}

// ---------------------------------------------------------------------------
// Kernel 4: fused flash attention with on-the-fly RPE bias MLP.
// One wave per (b, h, 16-row q tile). K processed 32 cols at a time.
// V tile staged to LDS transposed so P@V B-fragments are ds_load_b128.
// ---------------------------------------------------------------------------
__global__ void attn_kernel(const float* __restrict__ qn, const float* __restrict__ kn,
                            const float* __restrict__ vp, const float* __restrict__ coords,
                            const float* __restrict__ Wr1, const float* __restrict__ br1,
                            const float* __restrict__ Wr2, const float* __restrict__ br2,
                            float* __restrict__ out) {
  const int QT = N_ / 16;
  int qt = blockIdx.x % QT;
  int h  = (blockIdx.x / QT) % H_;
  int b  = blockIdx.x / (QT * H_);
  int lane = threadIdx.x & 31;
  int half = lane >> 4, l15 = lane & 15;
  int q0 = qt * 16;

  __shared__ unsigned short sP[16 * 32];     // P tile, bf16 [qrow][kv]
  __shared__ unsigned short sV[64 * LDB_];   // V tile, bf16, transposed [dh][kv]

  // RPE constants: uniform addresses -> scalar loads, kept in registers
  float w1a[16], w1b[16], c1r[16], w2[16];
#pragma unroll
  for (int j = 0; j < 16; ++j) {
    w1a[j] = Wr1[j];
    w1b[j] = Wr1[16 + j];
    c1r[j] = br1[j];
    w2[j]  = Wr2[j * H_ + h];
  }
  float b2 = br2[h];

  // q A-fragments (dh 0..31 and 32..63), A-layout row = lane&15
  v16bf aq[2];
  {
    const float* qrow = qn + (size_t)(b * N_ + q0 + l15) * D_ + h * DH_;
    int kb = half << 3;
#pragma unroll
    for (int ss = 0; ss < 2; ++ss) {
#pragma unroll
      for (int i = 0; i < 8; ++i) {
        aq[ss][i]     = (__bf16)qrow[ss * 32 + kb + i];
        aq[ss][8 + i] = (__bf16)qrow[ss * 32 + kb + 16 + i];
      }
    }
  }
  // coords of the 8 q rows this lane owns in C-layout (M = half*8 + r)
  float cqx[8], cqy[8];
#pragma unroll
  for (int r = 0; r < 8; ++r) {
    int qr = q0 + half * 8 + r;
    cqx[r] = coords[(size_t)(b * N_ + qr) * 2 + 0];
    cqy[r] = coords[(size_t)(b * N_ + qr) * 2 + 1];
  }

  float m[8], l[8];
#pragma unroll
  for (int r = 0; r < 8; ++r) { m[r] = -3.0e38f; l[r] = 0.f; }
  v8f O[4] = {};

  for (int kt = 0; kt < N_ / 32; ++kt) {
    int kbase = kt * 32;

    // ---- stage V tile (32 kv x 64 dh) -> LDS transposed [dh][kv] ----
    {
      const float* vrow = vp + (size_t)(b * N_ + kbase + lane) * D_ + h * DH_;
#pragma unroll
      for (int j4 = 0; j4 < 16; ++j4) {
        float4 f = *(const float4*)(vrow + j4 * 4);
        sV[(j4 * 4 + 0) * LDB_ + lane] = bfbits(f.x);
        sV[(j4 * 4 + 1) * LDB_ + lane] = bfbits(f.y);
        sV[(j4 * 4 + 2) * LDB_ + lane] = bfbits(f.z);
        sV[(j4 * 4 + 3) * LDB_ + lane] = bfbits(f.w);
      }
    }

    // ---- S = qn @ kn^T for 32 k-columns (two 16x16 tiles) ----
    v8f S0 = {}, S1 = {};
#pragma unroll
    for (int ss = 0; ss < 2; ++ss) {
      int dbase = ss * 32;
      const float* k0p = kn + (size_t)(b * N_ + kbase + l15) * D_ + h * DH_ + dbase + (half << 3);
      const float* k1p = kn + (size_t)(b * N_ + kbase + 16 + l15) * D_ + h * DH_ + dbase + (half << 3);
      v16bf b0, b1;
#pragma unroll
      for (int i = 0; i < 8; ++i) { b0[i] = (__bf16)k0p[i]; b1[i] = (__bf16)k1p[i]; b0[8 + i] = (__bf16)k0p[16 + i]; b1[8 + i] = (__bf16)k1p[16 + i]; }
      S0 = __builtin_amdgcn_wmma_f32_16x16x32_bf16(false, aq[ss], false, b0,
                                                   (short)0, S0, false, false);
      S1 = __builtin_amdgcn_wmma_f32_16x16x32_bf16(false, aq[ss], false, b1,
                                                   (short)0, S1, false, false);
    }

    // k-column coords for this lane's two columns
    float ckx0 = coords[(size_t)(b * N_ + kbase + l15) * 2 + 0];
    float cky0 = coords[(size_t)(b * N_ + kbase + l15) * 2 + 1];
    float ckx1 = coords[(size_t)(b * N_ + kbase + 16 + l15) * 2 + 0];
    float cky1 = coords[(size_t)(b * N_ + kbase + 16 + l15) * 2 + 1];

    // ---- bias MLP + online softmax ----
#pragma unroll
    for (int r = 0; r < 8; ++r) {
      float rx0 = cqx[r] - ckx0, ry0 = cqy[r] - cky0;
      float rx1 = cqx[r] - ckx1, ry1 = cqy[r] - cky1;
      float acc0 = b2, acc1 = b2;
#pragma unroll
      for (int j = 0; j < 16; ++j) {
        float h0 = fmaxf(rx0 * w1a[j] + ry0 * w1b[j] + c1r[j], 0.f);
        float h1 = fmaxf(rx1 * w1a[j] + ry1 * w1b[j] + c1r[j], 0.f);
        acc0 += h0 * w2[j];
        acc1 += h1 * w2[j];
      }
      float s0 = S0[r] + acc0;
      float s1 = S1[r] + acc1;

      float t = fmaxf(s0, s1);
      t = fmaxf(t, __shfl_xor(t, 1, 32));
      t = fmaxf(t, __shfl_xor(t, 2, 32));
      t = fmaxf(t, __shfl_xor(t, 4, 32));
      t = fmaxf(t, __shfl_xor(t, 8, 32));
      float mn    = fmaxf(m[r], t);
      float alpha = __expf(m[r] - mn);
      float p0 = __expf(s0 - mn), p1 = __expf(s1 - mn);
      float rsum = p0 + p1;
      rsum += __shfl_xor(rsum, 1, 32);
      rsum += __shfl_xor(rsum, 2, 32);
      rsum += __shfl_xor(rsum, 4, 32);
      rsum += __shfl_xor(rsum, 8, 32);
      l[r] = l[r] * alpha + rsum;
      m[r] = mn;
#pragma unroll
      for (int f = 0; f < 4; ++f) O[f][r] *= alpha;

      sP[(half * 8 + r) * 32 + l15]      = bfbits(p0);
      sP[(half * 8 + r) * 32 + 16 + l15] = bfbits(p1);
    }
    __syncthreads();

    // ---- re-read P as 16x32 A-fragment (C-layout -> A-layout transpose) ----
    BF16x16 pa;
    {
      const unsigned short* base = sP + l15 * 32 + (half << 3);
      uint4 lo = *(const uint4*)(base);        // K = kb..kb+7
      uint4 hi = *(const uint4*)(base + 16);   // K = kb+16..kb+23
      pa.u[0] = lo.x; pa.u[1] = lo.y; pa.u[2] = lo.z; pa.u[3] = lo.w;
      pa.u[4] = hi.x; pa.u[5] = hi.y; pa.u[6] = hi.z; pa.u[7] = hi.w;
    }

    // ---- O += P @ V (4 WMMAs, B-fragments from LDS) ----
#pragma unroll
    for (int f = 0; f < 4; ++f) {
      const unsigned short* vb = sV + (f * 16 + l15) * LDB_ + (half << 3);
      uint4 lo = *(const uint4*)(vb);
      uint4 hi = *(const uint4*)(vb + 16);
      BF16x16 bb;
      bb.u[0] = lo.x; bb.u[1] = lo.y; bb.u[2] = lo.z; bb.u[3] = lo.w;
      bb.u[4] = hi.x; bb.u[5] = hi.y; bb.u[6] = hi.z; bb.u[7] = hi.w;
      O[f] = __builtin_amdgcn_wmma_f32_16x16x32_bf16(false, pa.v, false, bb.v,
                                                     (short)0, O[f], false, false);
    }
    __syncthreads();
  }

  // ---- epilogue: divide by softmax denominator, write [B,N,D] ----
#pragma unroll
  for (int f = 0; f < 4; ++f) {
#pragma unroll
    for (int r = 0; r < 8; ++r) {
      int row = q0 + half * 8 + r;
      int col = h * DH_ + f * 16 + l15;
      out[(size_t)(b * N_ + row) * D_ + col] = O[f][r] / l[r];
    }
  }
}

// ---------------------------------------------------------------------------
// Kernel 5: out = LN(a + bres, g, bt)
// ---------------------------------------------------------------------------
__global__ void resln_kernel(const float* __restrict__ a, const float* __restrict__ bres,
                             const float* __restrict__ g, const float* __restrict__ bt,
                             float* __restrict__ out) {
  int row = blockIdx.x, tid = threadIdx.x;
  const float* ar = a + (size_t)row * D_;
  const float* br = bres + (size_t)row * D_;
  float t[2]; float s = 0.f, s2 = 0.f;
#pragma unroll
  for (int e = 0; e < 2; ++e) {
    int d = tid + e * 256;
    float v = ar[d] + br[d];
    t[e] = v; s += v; s2 += v * v;
  }
  __shared__ float rs[256], rs2[256];
  rs[tid] = s; rs2[tid] = s2; __syncthreads();
  for (int o = 128; o > 0; o >>= 1) {
    if (tid < o) { rs[tid] += rs[tid + o]; rs2[tid] += rs2[tid + o]; }
    __syncthreads();
  }
  float mean = rs[0] * (1.0f / D_);
  float var  = rs2[0] * (1.0f / D_) - mean * mean;
  float inv  = rsqrtf(var + 1e-5f);
#pragma unroll
  for (int e = 0; e < 2; ++e) {
    int d = tid + e * 256;
    out[(size_t)row * D_ + d] = (t[e] - mean) * inv * g[d] + bt[d];
  }
}

// ---------------------------------------------------------------------------
extern "C" void kernel_launch(void* const* d_in, const int* in_sizes, int n_in,
                              void* d_out, int out_size, void* d_ws, size_t ws_size,
                              hipStream_t stream) {
  (void)in_sizes; (void)n_in; (void)out_size; (void)ws_size;
  const float* x      = (const float*)d_in[0];
  const float* coords = (const float*)d_in[1];
  const float* Wq  = (const float*)d_in[2];  const float* bq  = (const float*)d_in[3];
  const float* Wk  = (const float*)d_in[4];  const float* bk  = (const float*)d_in[5];
  const float* Wv  = (const float*)d_in[6];  const float* bv  = (const float*)d_in[7];
  const float* Wo  = (const float*)d_in[8];  const float* bo  = (const float*)d_in[9];
  const float* ls  = (const float*)d_in[10];
  const float* Wpe = (const float*)d_in[11]; const float* bpe = (const float*)d_in[12];
  const float* Wr1 = (const float*)d_in[13]; const float* br1 = (const float*)d_in[14];
  const float* Wr2 = (const float*)d_in[15]; const float* br2 = (const float*)d_in[16];
  const float* gpeq = (const float*)d_in[17]; const float* btpeq = (const float*)d_in[18];
  const float* gpek = (const float*)d_in[19]; const float* btpek = (const float*)d_in[20];
  const float* g2  = (const float*)d_in[21]; const float* bt2 = (const float*)d_in[22];
  const float* g3  = (const float*)d_in[23]; const float* bt3 = (const float*)d_in[24];
  const float* Wff1 = (const float*)d_in[25];
  const float* Wff2 = (const float*)d_in[26];
  float* out = (float*)d_out;

  float* ws = (float*)d_ws;
  const size_t MD = (size_t)M_ * D_;
  float* qin = ws;                 // later reused as att_out
  float* kin = ws + MD;            // later reused as proj
  float* qp  = ws + 2 * MD;
  float* kp  = ws + 3 * MD;
  float* vp  = ws + 4 * MD;
  float* x1  = ws + 5 * MD;
  float* hff = ws + 6 * MD;                         // M * FF
  float* mlp = ws + 6 * MD + (size_t)M_ * FF_;

  const int gemmD  = (M_ / BM_) * (D_ / BN_);       // 256 blocks
  const int gemmFF = (M_ / BM_) * (FF_ / BN_);      // 1024 blocks

  // 1. PE add + twin LayerNorm
  pe_ln2_kernel<<<M_, 256, 0, stream>>>(x, coords, Wpe, bpe, gpeq, btpeq, gpek, btpek, qin, kin);
  // 2. QKV projections (WMMA)
  gemm_wmma_kernel<<<gemmD, 256, 0, stream>>>(qin, Wq, bq, qp, M_, D_, D_, 0);
  gemm_wmma_kernel<<<gemmD, 256, 0, stream>>>(kin, Wk, bk, kp, M_, D_, D_, 0);
  gemm_wmma_kernel<<<gemmD, 256, 0, stream>>>(x,   Wv, bv, vp, M_, D_, D_, 0);
  // 3. cosine normalization; logit scale folded into q
  headnorm_kernel<<<M_ * H_, 64, 0, stream>>>(qp, ls, 1);
  headnorm_kernel<<<M_ * H_, 64, 0, stream>>>(kp, ls, 0);
  // 4. fused flash attention + RPE bias (WMMA), writes att_out into qin
  attn_kernel<<<B_ * H_ * (N_ / 16), 32, 0, stream>>>(qp, kp, vp, coords,
                                                      Wr1, br1, Wr2, br2, qin);
  // 5. output projection (WMMA) -> proj in kin
  gemm_wmma_kernel<<<gemmD, 256, 0, stream>>>(qin, Wo, bo, kin, M_, D_, D_, 0);
  // 6. x1 = LN(x + proj)
  resln_kernel<<<M_, 256, 0, stream>>>(x, kin, g2, bt2, x1);
  // 7. MLP with fused LeakyReLU(0.2) (WMMA)
  gemm_wmma_kernel<<<gemmFF, 256, 0, stream>>>(x1, Wff1, nullptr, hff, M_, D_, FF_, 1);
  gemm_wmma_kernel<<<gemmD, 256, 0, stream>>>(hff, Wff2, nullptr, mlp, M_, FF_, D_, 1);
  // 8. final LN -> d_out
  resln_kernel<<<M_, 256, 0, stream>>>(x1, mlp, g3, bt3, out);
}
